// FlashRAMLAAttention_48799418417581
// MI455X (gfx1250) — matrix operand order repaired and checked by reading, not compiled, not run on hardware
//
#include <hip/hip_runtime.h>

// ---------------------------------------------------------------------------
// MLA attention forward for MI455X (gfx1250), bf16 WMMA pipeline + TDM staging.
// B=2, T=1024, E=2048, H=16, D=128, L=512.
// ---------------------------------------------------------------------------

typedef __bf16 bf16_t;
typedef __attribute__((ext_vector_type(16))) __bf16       v16bf;
typedef __attribute__((ext_vector_type(8)))  float        v8f;
typedef __attribute__((ext_vector_type(4)))  unsigned int v4u;
typedef __attribute__((ext_vector_type(4)))  unsigned int u32x4;
typedef __attribute__((ext_vector_type(8)))  int          i32x8;
typedef __attribute__((ext_vector_type(4)))  int          i32x4;

static __device__ __forceinline__ v8f zero8() {
  v8f z;
#pragma unroll
  for (int i = 0; i < 8; ++i) z[i] = 0.0f;
  return z;
}

static __device__ __forceinline__ v8f wmma_bf16(v16bf a, v16bf b, v8f c) {
  // D = A(16x32 bf16) * B(32x16 bf16) + C(16x16 f32)
  return __builtin_amdgcn_wmma_f32_16x16x32_bf16(
      /*neg_a=*/false, a, /*neg_b=*/false, b,
      /*c_mod=*/(short)0, c, /*reuse_a=*/false, /*reuse_b=*/false);
}

// A-matrix fragment (16x32, MxK), row-major source, `base` at (m0,k0), ld in elems.
// ISA layout: lane l: M = l%16; K = {0..7,16..23} (l<16) or {8..15,24..31}.
static __device__ __forceinline__ v16bf load_a_frag(const bf16_t* base, int ld) {
  const int l   = threadIdx.x & 31;
  const int row = l & 15;
  const int kb  = (l < 16) ? 0 : 8;
  const bf16_t* p = base + (long)row * ld + kb;
  union { v4u u[2]; v16bf v; } f;
  f.u[0] = *(const v4u*)(p);
  f.u[1] = *(const v4u*)(p + 16);
  return f.v;
}

// B-matrix fragment (32x16, KxN) from transposed storage BT[N][K], `base` at (n0,k0).
// ISA layout: lane l: N = l%16; K = {0..7,16..23} (l<16) or {8..15,24..31}.
static __device__ __forceinline__ v16bf load_bt_frag(const bf16_t* base, long ld) {
  const int l  = threadIdx.x & 31;
  const int n  = l & 15;
  const int kb = (l < 16) ? 0 : 8;
  const bf16_t* p = base + (long)n * ld + kb;
  union { v4u u[2]; v16bf v; } f;
  f.u[0] = *(const v4u*)(p);
  f.u[1] = *(const v4u*)(p + 16);
  return f.v;
}

// P fragment from per-wave LDS tile (16 rows x 32 cols bf16, row stride 40 elems).
static __device__ __forceinline__ v16bf load_p_frag(const bf16_t* base) {
  const int l = threadIdx.x & 31;
  const int m = l & 15;
  const unsigned int* row = (const unsigned int*)(base + m * 40);
  const int o = (l < 16) ? 0 : 4;
  union { unsigned int u[8]; v16bf v; } f;
#pragma unroll
  for (int j = 0; j < 4; ++j) {
    f.u[j]     = row[o + j];       // K = {0..7} or {8..15}
    f.u[4 + j] = row[8 + o + j];   // K = {16..23} or {24..31}
  }
  return f.v;
}

// ---------------------------------------------------------------------------
// Tensor Data Mover: DMA a [tile_d1 rows x tile_d0 elems] bf16 tile (row
// stride `stride0` elems in global) into LDS at byte offset `lds_off`.
// pad_enable with interval=16 DWORDs, amount=4 DWORDs -> LDS row stride
// becomes 32 elems + 8 pad = 40 elems, matching load_bt_frag(ld=40).
// ---------------------------------------------------------------------------
static __device__ __forceinline__ void tdm_load_tile_bf16(
    unsigned lds_off, const bf16_t* gaddr,
    unsigned tile_d0, unsigned tile_d1, unsigned long stride0) {
  const unsigned long ga = (unsigned long)gaddr;
  u32x4 g0;
  g0[0] = 1u;                                   // count=1, user descriptor
  g0[1] = lds_off;                              // lds_addr (bytes)
  g0[2] = (unsigned)ga;                         // global_addr[31:0]
  g0[3] = (unsigned)(ga >> 32) | (2u << 30);    // global_addr[56:32] | type=2
  i32x8 g1;
  // wg_mask=0 | data_size=1(2B)<<16 | pad_enable<<20 | pad_interval=3(16dw)<<22
  // | pad_amount=3(4dw)<<25
  g1[0] = (int)((1u << 16) | (1u << 20) | (3u << 22) | (3u << 25));
  g1[1] = (int)((tile_d0 & 0xFFFFu) << 16);                     // tensor_dim0 lo16 @ [63:48]
  g1[2] = (int)((tile_d0 >> 16) | ((tile_d1 & 0xFFFFu) << 16)); // dim0 hi16 | dim1 lo16
  g1[3] = (int)((tile_d1 >> 16) | (tile_d0 << 16));             // dim1 hi16 | tile_dim0
  g1[4] = (int)(tile_d1 & 0xFFFFu);                             // tile_dim1 | tile_dim2=0
  g1[5] = (int)(unsigned)(stride0 & 0xFFFFFFFFu);               // dim0_stride[31:0]
  g1[6] = (int)(unsigned)((stride0 >> 32) & 0xFFFFu);           // dim0_stride[47:32]
  g1[7] = 0;
  i32x4 gz4;
  gz4[0] = 0; gz4[1] = 0; gz4[2] = 0; gz4[3] = 0;
  i32x8 gz8;
#pragma unroll
  for (int i = 0; i < 8; ++i) gz8[i] = 0;
  // clang-23 / therock form: 6 args (g0, g1, g2, g3, g4, cpol).
  __builtin_amdgcn_tensor_load_to_lds(g0, g1, gz4, gz4, gz8, 0);
}

// ---------------------------------------------------------------------------
// Elementwise cast fp32 -> bf16
// ---------------------------------------------------------------------------
__global__ void cast_bf16_kernel(const float* __restrict__ in,
                                 bf16_t* __restrict__ out, long n) {
  long i = (long)blockIdx.x * blockDim.x + threadIdx.x;
  if (i < n) out[i] = (bf16_t)in[i];
}

// Transposing cast: in[R][C] fp32 -> out[C][R] bf16 (batched over blockIdx.z).
__global__ void transpose_cast_kernel(const float* __restrict__ in,
                                      bf16_t* __restrict__ out, int R, int C) {
  const long base = (long)blockIdx.z * R * C;
  const int c = blockIdx.x * 32 + (threadIdx.x & 31);
  const int r = blockIdx.y * 8 + (threadIdx.x >> 5);
  if (r < R && c < C)
    out[base + (long)c * R + r] = (bf16_t)in[base + (long)r * C + c];
}

// ---------------------------------------------------------------------------
// Generic bf16 WMMA GEMM: C[M][N] = A[M][K] * BT[N][K]^T
// 128 threads (4 waves), 64x64 block tile.
// BT tiles are DMA'd into double-buffered LDS by the Tensor Data Mover
// (wave 0 drives the descriptors; TENSORcnt keeps one DMA in flight under
// the WMMAs of the current step).
// OUT_F32: store fp32 instead of bf16.  STORE_T: store C transposed (C[n][m]).
// M,N multiples of 64; K multiple of 32 (true for all uses here).
// ---------------------------------------------------------------------------
template <bool OUT_F32, bool STORE_T>
__global__ __launch_bounds__(128) void gemm_bt_kernel(
    const bf16_t* __restrict__ A, int lda, long sA,
    const bf16_t* __restrict__ BT, int ldbt, long sB,
    void* __restrict__ Cv, int ldc, long sC,
    int M, int N, int K) {
  __shared__ bf16_t bs[2][64 * 40];  // double-buffered 64 x 32 tile (stride 40)

  const int z = blockIdx.z;
  A  += (long)z * sA;
  BT += (long)z * sB;
  const long cbase = (long)z * sC;

  const int nbase = blockIdx.x * 64;
  const int mbase = blockIdx.y * 64;
  const int wid   = threadIdx.x >> 5;
  const int lane  = threadIdx.x & 31;
  const int mrow  = mbase + wid * 16;

  const bf16_t* Btile = BT + (long)nbase * ldbt;  // (nbase, 0)

  v8f acc[4];
#pragma unroll
  for (int i = 0; i < 4; ++i) acc[i] = zero8();

  // Prologue: DMA first BT tile into buffer 0.
  if (wid == 0) {
    tdm_load_tile_bf16((unsigned)(unsigned long)&bs[0][0], Btile, 32, 64,
                       (unsigned long)ldbt);
  }

  for (int k0 = 0; k0 < K; k0 += 32) {
    const int cur = (k0 >> 5) & 1;
    if (wid == 0) {
      if (k0 + 32 < K) {
        // Kick off DMA for the next tile into the other buffer, then wait
        // until only that one is outstanding (=> current tile landed).
        tdm_load_tile_bf16((unsigned)(unsigned long)&bs[cur ^ 1][0],
                           Btile + (k0 + 32), 32, 64, (unsigned long)ldbt);
        __builtin_amdgcn_s_wait_tensorcnt(1);
      } else {
        __builtin_amdgcn_s_wait_tensorcnt(0);
      }
    }
    __syncthreads();  // publish DMA'd tile to all waves

    v16bf af = load_a_frag(A + (long)mrow * lda + k0, lda);
#pragma unroll
    for (int nt = 0; nt < 4; ++nt) {
      v16bf bfr = load_bt_frag(&bs[cur][nt * 16 * 40], 40);
      acc[nt] = wmma_bf16(af, bfr, acc[nt]);
    }
    __syncthreads();  // drain reads of bs[cur] before TDM overwrites it (k0+64)
  }

  // Store: lane l holds D[r + 8*(l/16)][l%16] per N-tile.
  const int chalf = (lane < 16) ? 0 : 8;
#pragma unroll
  for (int nt = 0; nt < 4; ++nt) {
    const int n = nbase + nt * 16 + (lane & 15);
#pragma unroll
    for (int r = 0; r < 8; ++r) {
      const int m = mrow + r + chalf;
      const float v = acc[nt][r];
      const long idx = STORE_T ? ((long)n * ldc + m) : ((long)m * ldc + n);
      if (OUT_F32) ((float*)Cv)[cbase + idx] = v;
      else         ((bf16_t*)Cv)[cbase + idx] = (bf16_t)v;
    }
  }
}

// ---------------------------------------------------------------------------
// Flash attention over the latent dim.
//   Qlat : [H][BTr][L]   (bf16)   BTr = B*T
//   Klat : [BTr][L]      (bf16)
//   VhT  : [H][D][BTr]   (bf16)   (V_lat @ v_up[h], stored transposed)
//   O    : [BTr][H*D]    (bf16)
// grid (T/64, H, B), 128 threads (4 waves); each wave owns 16 query rows.
// ---------------------------------------------------------------------------
__global__ __launch_bounds__(128) void flash_mla_kernel(
    const bf16_t* __restrict__ Qlat, const bf16_t* __restrict__ Klat,
    const bf16_t* __restrict__ VhT, bf16_t* __restrict__ O,
    int T, int BTr) {
  const int L = 512, D = 128, E = 2048;
  __shared__ bf16_t ps[4][16 * 40];  // per-wave P tile (16x32, padded)

  const int wid  = threadIdx.x >> 5;
  const int lane = threadIdx.x & 31;
  const int b = blockIdx.z;
  const int h = blockIdx.y;
  const int qbase = blockIdx.x * 64 + wid * 16;
  const long rowQ = (long)b * T + qbase;

  const bf16_t* Qbase = Qlat + (long)h * BTr * L + rowQ * L;
  const bf16_t* Vbase = VhT + (long)h * D * BTr + (long)b * T;  // col offset b*T

  // Preload all Q_lat fragments for this wave's 16 rows (L/32 = 16 frags).
  v16bf qf[16];
#pragma unroll
  for (int i = 0; i < 16; ++i) qf[i] = load_a_frag(Qbase + i * 32, L);

  v8f oacc[8];
#pragma unroll
  for (int i = 0; i < 8; ++i) oacc[i] = zero8();
  float mst[8], lst[8];
#pragma unroll
  for (int r = 0; r < 8; ++r) { mst[r] = -__builtin_inff(); lst[r] = 0.0f; }

  const float scale = 0.044194173824159216f;  // 512^-0.5
  const int rhalf = (lane < 16) ? 0 : 8;
  const int qmax = qbase + 15;
  bf16_t* pw = &ps[wid][0];

  for (int kb = 0; kb <= qmax; kb += 32) {
    const bf16_t* Kb0 = Klat + ((long)b * T + kb) * L;
    __builtin_prefetch(Kb0 + 32L * L, 0, 0);  // stream next key chunk

    // S tiles: 16 queries x 32 keys, K-reduction over L=512.
    v8f s0 = zero8(), s1 = zero8();
#pragma unroll
    for (int kk = 0; kk < 16; ++kk) {
      v16bf b0 = load_bt_frag(Kb0 + kk * 32, L);
      s0 = wmma_bf16(qf[kk], b0, s0);
      v16bf b1 = load_bt_frag(Kb0 + (long)16 * L + kk * 32, L);
      s1 = wmma_bf16(qf[kk], b1, s1);
    }

    // Causal mask + scale + row max (16-lane half-wave reductions).
    const int col0 = kb + (lane & 15);
    const int col1 = col0 + 16;
    float rmax[8];
#pragma unroll
    for (int r = 0; r < 8; ++r) {
      const int rowg = qbase + rhalf + r;
      float a = s0[r] * scale;
      float c = s1[r] * scale;
      if (col0 > rowg) a = -__builtin_inff();
      if (col1 > rowg) c = -__builtin_inff();
      s0[r] = a; s1[r] = c;
      float mx = fmaxf(a, c);
      mx = fmaxf(mx, __shfl_xor(mx, 1));
      mx = fmaxf(mx, __shfl_xor(mx, 2));
      mx = fmaxf(mx, __shfl_xor(mx, 4));
      mx = fmaxf(mx, __shfl_xor(mx, 8));
      rmax[r] = mx;
    }

    // Online softmax update; write P (bf16) into per-wave LDS tile.
#pragma unroll
    for (int r = 0; r < 8; ++r) {
      const float mn = fmaxf(mst[r], rmax[r]);
      const float alpha = __expf(mst[r] - mn);
      mst[r] = mn;
      const float p0 = __expf(s0[r] - mn);
      const float p1 = __expf(s1[r] - mn);
      float rs = p0 + p1;
      rs += __shfl_xor(rs, 1);
      rs += __shfl_xor(rs, 2);
      rs += __shfl_xor(rs, 4);
      rs += __shfl_xor(rs, 8);
      lst[r] = lst[r] * alpha + rs;
#pragma unroll
      for (int nt = 0; nt < 8; ++nt) oacc[nt][r] *= alpha;
      const int rr = r + rhalf;
      pw[rr * 40 + (lane & 15)]      = (bf16_t)p0;
      pw[rr * 40 + 16 + (lane & 15)] = (bf16_t)p1;
    }
    __asm__ volatile("s_wait_dscnt 0" ::: "memory");

    // O += P(16x32) @ Vh(32x128): 8 N-tiles.
    v16bf pf = load_p_frag(pw);
#pragma unroll
    for (int nt = 0; nt < 8; ++nt) {
      v16bf vf = load_bt_frag(Vbase + (long)(nt * 16) * BTr + kb, BTr);
      oacc[nt] = wmma_bf16(pf, vf, oacc[nt]);
    }
  }

  // Normalize and store O[b*T + m][h*D + n] as bf16.
#pragma unroll
  for (int r = 0; r < 8; ++r) {
    const float inv = 1.0f / lst[r];
    const int m = qbase + rhalf + r;
    const long orow = ((long)b * T + m) * E + (long)h * D;
#pragma unroll
    for (int nt = 0; nt < 8; ++nt)
      O[orow + nt * 16 + (lane & 15)] = (bf16_t)(oacc[nt][r] * inv);
  }
}

// ---------------------------------------------------------------------------
// Host-side orchestration
// ---------------------------------------------------------------------------
extern "C" void kernel_launch(void* const* d_in, const int* in_sizes, int n_in,
                              void* d_out, int out_size, void* d_ws, size_t ws_size,
                              hipStream_t stream) {
  (void)in_sizes; (void)n_in; (void)out_size; (void)ws_size;
  const int B = 2, T = 1024, E = 2048, H = 16, D = 128, L = 512;
  const int BTr = B * T;  // 2048

  const float* x    = (const float*)d_in[0];
  const float* Wq   = (const float*)d_in[1];
  const float* Wk   = (const float*)d_in[2];
  const float* Wv   = (const float*)d_in[3];
  const float* q2l  = (const float*)d_in[4];
  const float* vup  = (const float*)d_in[5];
  const float* Wout = (const float*)d_in[6];
  float* out = (float*)d_out;

  // Workspace carve-up (bf16 elements).
  bf16_t* w = (bf16_t*)d_ws;
  bf16_t* xb    = w; w += (long)BTr * E;       // [2048][2048]
  bf16_t* WqT   = w; w += (long)E * E;         // [E][E]
  bf16_t* WkT   = w; w += (long)L * E;         // [L][E]
  bf16_t* WvT   = w; w += (long)L * E;         // [L][E]
  bf16_t* q2lT  = w; w += (long)H * L * D;     // [H][L][D]
  bf16_t* vupT  = w; w += (long)H * D * L;     // [H][D][L]
  bf16_t* WoutT = w; w += (long)E * E;         // [E][E]
  bf16_t* Qb    = w; w += (long)BTr * E;       // [2048][2048]
  bf16_t* Kb    = w; w += (long)BTr * L;       // [2048][512]
  bf16_t* Vb    = w; w += (long)BTr * L;       // [2048][512]
  bf16_t* QLat  = w; w += (long)H * BTr * L;   // [H][2048][512]
  bf16_t* VhT   = w; w += (long)H * D * BTr;   // [H][128][2048]
  bf16_t* Ob    = w; w += (long)BTr * E;       // [2048][2048]

  // 1) Casts / transposes to bf16.
  {
    long n = (long)BTr * E;
    cast_bf16_kernel<<<dim3((unsigned)((n + 255) / 256)), 256, 0, stream>>>(x, xb, n);
  }
  transpose_cast_kernel<<<dim3(E / 32, E / 8, 1), 256, 0, stream>>>(Wq, WqT, E, E);
  transpose_cast_kernel<<<dim3(L / 32, E / 8, 1), 256, 0, stream>>>(Wk, WkT, E, L);
  transpose_cast_kernel<<<dim3(L / 32, E / 8, 1), 256, 0, stream>>>(Wv, WvT, E, L);
  transpose_cast_kernel<<<dim3(L / 32, D / 8, H), 256, 0, stream>>>(q2l, q2lT, D, L);
  transpose_cast_kernel<<<dim3(D / 32, L / 8, H), 256, 0, stream>>>(vup, vupT, L, D);
  transpose_cast_kernel<<<dim3(E / 32, E / 8, 1), 256, 0, stream>>>(Wout, WoutT, E, E);

  // 2) Q = x @ Wq                        [2048,2048] x [2048,2048]
  gemm_bt_kernel<false, false><<<dim3(E / 64, BTr / 64, 1), 128, 0, stream>>>(
      xb, E, 0, WqT, E, 0, Qb, E, 0, BTr, E, E);
  // 3) K_lat = x @ Wk_down, V_lat = x @ Wv_down
  gemm_bt_kernel<false, false><<<dim3(L / 64, BTr / 64, 1), 128, 0, stream>>>(
      xb, E, 0, WkT, E, 0, Kb, L, 0, BTr, L, E);
  gemm_bt_kernel<false, false><<<dim3(L / 64, BTr / 64, 1), 128, 0, stream>>>(
      xb, E, 0, WvT, E, 0, Vb, L, 0, BTr, L, E);
  // 4) Q_lat[h] = Q[:, h*D:(h+1)*D] @ q_to_latent[h]   (batched over heads)
  gemm_bt_kernel<false, false><<<dim3(L / 64, BTr / 64, H), 128, 0, stream>>>(
      Qb, E, (long)D, q2lT, D, (long)L * D, QLat, L, (long)BTr * L, BTr, L, D);
  // 5) Vh[h]^T = (V_lat @ v_up[h])^T   (transposed store -> [D][tokens])
  gemm_bt_kernel<false, true><<<dim3(D / 64, BTr / 64, H), 128, 0, stream>>>(
      Vb, L, 0, vupT, L, (long)D * L, VhT, BTr, (long)D * BTr, BTr, D, L);

  // 6) Flash attention (causal, online softmax) -> Ob [B*T][H*D]
  flash_mla_kernel<<<dim3(T / 64, H, B), 128, 0, stream>>>(QLat, Kb, VhT, Ob, T, BTr);

  // 7) out = Ob @ Wout  (fp32 output)
  gemm_bt_kernel<true, false><<<dim3(E / 64, BTr / 64, 1), 128, 0, stream>>>(
      Ob, E, 0, WoutT, E, 0, out, E, 0, BTr, E, E);
}
